// Model_29308856828459
// MI455X (gfx1250) — compile-verified, matrix operand order checked by reading, not run on hardware
//
#include <hip/hip_runtime.h>
#include <hip/hip_bf16.h>

// ---------------------------------------------------------------------------
// Vanilla tanh RNN for MI455X (gfx1250).
//   SEQ=512, BATCH=64, IN=1024, HID=1024, fp32 in/out.
// Strategy:
//   Pass A: convert x, W_ih, W_hh to bf16 (ws), bias = b_ih + b_hh.
//   Pass B: xp = x @ W_ih^T + bias via v_wmma_f32_16x16x32_bf16, written to
//           d_out (xp slot == final output slot).
//   Pass C: persistent single-workgroup scan (32 wave32s on one WGP),
//           h double-buffered in LDS (2 x 64 x (1024+8) bf16 = 258 KB of the
//           320 KB WGP LDS), one barrier per timestep.  Each step updates
//           d_out[s] in place: out = tanh(xp + h @ W_hh^T).
// ---------------------------------------------------------------------------

#define RNN_SEQ   512
#define RNN_B     64
#define RNN_K     1024
#define RNN_H     1024
#define RNN_M     (RNN_SEQ * RNN_B)   // 32768
#define LDS_LDH   (RNN_H + 8)         // pad 8 halves -> 2064B stride, bank-safe

typedef __bf16 v16bf __attribute__((ext_vector_type(16)));
typedef float  v8f   __attribute__((ext_vector_type(8)));

struct alignas(32) BFrag {
    union {
        v16bf bf;
        uint4 q[2];
    };
};

__device__ __forceinline__ unsigned short f32_to_bf16_rne(float f) {
    unsigned int u = __float_as_uint(f);
    u += 0x7FFFu + ((u >> 16) & 1u);       // round to nearest even
    return (unsigned short)(u >> 16);
}

// Load a 16x32 bf16 fragment (A layout; B uses the same pattern with N as the
// "row" since both operands are row-major along K).  `base` points at
// (row0, k0); `ld` is the leading dimension in elements (multiple of 8).
// ISA 7.12.2: lanes 0-15 -> rows, halves 0-7 -> K {0..7,16..23} (lo half-wave)
// or K {8..15,24..31} (hi half-wave): two contiguous 16B chunks per lane.
__device__ __forceinline__ v16bf load_frag(const unsigned short* base, int ld) {
    const int lane = threadIdx.x & 31;
    const int l15  = lane & 15;
    const int hi   = lane >> 4;
    const unsigned short* p = base + (size_t)l15 * ld + hi * 8;
    BFrag f;
    f.q[0] = *(const uint4*)(p);
    f.q[1] = *(const uint4*)(p + 16);
    return f.bf;
}

__device__ __forceinline__ v8f wmma_bf16(v16bf a, v16bf b, v8f c) {
    return __builtin_amdgcn_wmma_f32_16x16x32_bf16(
        /*neg_a=*/false, a, /*neg_b=*/false, b,
        /*c_mod=*/(short)0, c, /*reuse_a=*/false, /*reuse_b=*/false);
}

// ---------------------------------------------------------------------------
// Pass A kernels
// ---------------------------------------------------------------------------
__global__ void cvt_f32_bf16(const float* __restrict__ in,
                             unsigned short* __restrict__ out, int n) {
    for (int i = blockIdx.x * blockDim.x + threadIdx.x; i < n;
         i += gridDim.x * blockDim.x)
        out[i] = f32_to_bf16_rne(in[i]);
}

__global__ void bias_sum(const float* __restrict__ b0,
                         const float* __restrict__ b1,
                         float* __restrict__ o) {
    int i = blockIdx.x * blockDim.x + threadIdx.x;
    if (i < RNN_H) o[i] = b0[i] + b1[i];
}

// ---------------------------------------------------------------------------
// Pass B: xp = x @ W_ih^T + bias   (M=32768, N=1024, K=1024)
// Block = 256 threads (8 waves, 4x2); block tile 128x64; wave tile 32x32.
// ---------------------------------------------------------------------------
__global__ void __launch_bounds__(256)
gemm_xp(const unsigned short* __restrict__ xb,
        const unsigned short* __restrict__ wih,
        const float* __restrict__ bias,
        float* __restrict__ out) {
    const int wave = threadIdx.x >> 5;
    const int wm   = wave >> 1;           // 0..3
    const int wn   = wave & 1;            // 0..1
    const int m0   = blockIdx.x * 128 + wm * 32;
    const int n0   = blockIdx.y * 64  + wn * 32;

    v8f acc00 = {}, acc01 = {}, acc10 = {}, acc11 = {};
    for (int k0 = 0; k0 < RNN_K; k0 += 32) {
        v16bf a0 = load_frag(xb  + (size_t)m0        * RNN_K + k0, RNN_K);
        v16bf a1 = load_frag(xb  + (size_t)(m0 + 16) * RNN_K + k0, RNN_K);
        v16bf b0 = load_frag(wih + (size_t)n0        * RNN_K + k0, RNN_K);
        v16bf b1 = load_frag(wih + (size_t)(n0 + 16) * RNN_K + k0, RNN_K);
        acc00 = wmma_bf16(a0, b0, acc00);
        acc01 = wmma_bf16(a0, b1, acc01);
        acc10 = wmma_bf16(a1, b0, acc10);
        acc11 = wmma_bf16(a1, b1, acc11);
    }

    const int lane = threadIdx.x & 31;
    const int l15  = lane & 15;
    const int hi   = lane >> 4;
    v8f accs[4] = {acc00, acc01, acc10, acc11};
#pragma unroll
    for (int t = 0; t < 4; ++t) {
        const int tm = m0 + (t >> 1) * 16;
        const int tn = n0 + (t & 1)  * 16;
        const int n  = tn + l15;
        const float bv = bias[n];
#pragma unroll
        for (int r = 0; r < 8; ++r) {
            const int m = tm + r + hi * 8;
            out[(size_t)m * RNN_H + n] = accs[t][r] + bv;
        }
    }
}

// ---------------------------------------------------------------------------
// Pass C: persistent scan.  One workgroup, 1024 threads = 32 waves.
// LDS: two h buffers [64][LDS_LDH] bf16.  Per step: 256 tiles (4x64), 8/wave;
// A-frags from LDS (ds_load_b128), B-frags (W_hh, 2MB bf16) from L2.
// ---------------------------------------------------------------------------
__global__ void __launch_bounds__(1024)
rnn_scan(const unsigned short* __restrict__ whh,
         float* __restrict__ out) {
    extern __shared__ unsigned short lds[];
    unsigned short* hbuf0 = lds;
    unsigned short* hbuf1 = lds + (size_t)RNN_B * LDS_LDH;

    for (int i = threadIdx.x; i < RNN_B * LDS_LDH; i += blockDim.x) {
        hbuf0[i] = 0;
        hbuf1[i] = 0;
    }
    __syncthreads();

    const int wave = threadIdx.x >> 5;     // 0..31
    const int lane = threadIdx.x & 31;
    const int l15  = lane & 15;
    const int hi   = lane >> 4;

    unsigned short* hc = hbuf0;            // h_{s}   (read)
    unsigned short* hn = hbuf1;            // h_{s+1} (write)

    for (int s = 0; s < RNN_SEQ; ++s) {
        float* outs = out + (size_t)s * RNN_B * RNN_H;
#pragma unroll 1
        for (int i = 0; i < 8; ++i) {
            const int t  = wave * 8 + i;   // 0..255
            const int mt = t >> 6;         // 0..3   (batch tile)
            const int nt = t & 63;         // 0..63  (hidden tile)
            const int m0 = mt * 16;
            const int n0 = nt * 16;

            v8f acc = {};
            for (int k0 = 0; k0 < RNN_H; k0 += 32) {
                v16bf a = load_frag(hc + (size_t)m0 * LDS_LDH + k0, LDS_LDH);
                v16bf b = load_frag(whh + (size_t)n0 * RNN_K + k0, RNN_K);
                acc = wmma_bf16(a, b, acc);
            }

            const int n = n0 + l15;
#pragma unroll
            for (int r = 0; r < 8; ++r) {
                const int m = m0 + r + hi * 8;
                const size_t idx = (size_t)m * RNN_H + n;
                const float v = tanhf(outs[idx] + acc[r]);  // xp_t + h W^T
                outs[idx] = v;                              // emit output
                hn[(size_t)m * LDS_LDH + n] = f32_to_bf16_rne(v);
            }
        }
        __syncthreads();                   // hn complete; hc free to reuse
        unsigned short* tmp = hc; hc = hn; hn = tmp;
    }
}

// ---------------------------------------------------------------------------
extern "C" void kernel_launch(void* const* d_in, const int* in_sizes, int n_in,
                              void* d_out, int out_size, void* d_ws, size_t ws_size,
                              hipStream_t stream) {
    const float* x    = (const float*)d_in[0];   // [512,64,1024]
    const float* W_ih = (const float*)d_in[1];   // [1024,1024]
    const float* W_hh = (const float*)d_in[2];   // [1024,1024]
    const float* b_ih = (const float*)d_in[3];   // [1024]
    const float* b_hh = (const float*)d_in[4];   // [1024]
    float* out = (float*)d_out;                  // [32768,1024]

    // Workspace layout (bytes):
    //   [0, 64MB)      x in bf16
    //   [64MB, 66MB)   W_ih bf16
    //   [66MB, 68MB)   W_hh bf16
    //   [68MB, +4KB)   bias sum f32
    char* ws = (char*)d_ws;
    unsigned short* xb    = (unsigned short*)(ws);
    unsigned short* wih_b = (unsigned short*)(ws + (size_t)RNN_M * RNN_K * 2);
    unsigned short* whh_b = (unsigned short*)(ws + (size_t)RNN_M * RNN_K * 2
                                                 + (size_t)RNN_H * RNN_K * 2);
    float* bias = (float*)(ws + (size_t)RNN_M * RNN_K * 2
                              + 2 * (size_t)RNN_H * RNN_K * 2);

    // Pass A: conversions.
    cvt_f32_bf16<<<4096, 256, 0, stream>>>(x, xb, RNN_M * RNN_K);
    cvt_f32_bf16<<<1024, 256, 0, stream>>>(W_ih, wih_b, RNN_H * RNN_K);
    cvt_f32_bf16<<<1024, 256, 0, stream>>>(W_hh, whh_b, RNN_H * RNN_K);
    bias_sum<<<4, 256, 0, stream>>>(b_ih, b_hh, bias);

    // Pass B: xp -> d_out.
    dim3 grid(RNN_M / 128, RNN_H / 64);
    gemm_xp<<<grid, 256, 0, stream>>>(xb, wih_b, bias, out);

    // Pass C: sequential scan, single persistent workgroup.
    const size_t lds_bytes = 2ull * RNN_B * LDS_LDH * sizeof(unsigned short);
    rnn_scan<<<1, 1024, lds_bytes, stream>>>(whh_b, out);
}
